// LineVectorizer_27797028339771
// MI455X (gfx1250) — compile-verified, matrix operand order checked by reading, not run on hardware
//
#include <hip/hip_runtime.h>
#include <hip/hip_bf16.h>
#include <stdint.h>

// ---------------------------------------------------------------------------
// LOI head for MI455X (gfx1250, wave32, WMMA, async-to-LDS double buffering).
//   fc1   : [B,256,HW] x [128,256]^T -> x_t [B, HW, 128]  (bf16, pixel-major)
//   gather: bilinear sample 32 pts/line, maxpool(4) -> feats [20000, 1024] bf16
//   mlp   : feats @ w1^T (relu) -> h1 ; h1 @ w2^T (relu) -> h2 ; h2 @ w3^T -> out
// GEMMs use v_wmma_f32_16x16x32_bf16 (f32 accumulate).
// ---------------------------------------------------------------------------

typedef __bf16 bf16_t;
typedef __attribute__((ext_vector_type(16))) __bf16 v16bf;
typedef __attribute__((ext_vector_type(8)))  float  v8f;

#define DIM_B    4
#define DIM_L    5000
#define CIN      256
#define HW       65536      // 256*256
#define DIM_LOI  128
#define DIM_FC   1024
#define NLINES   (DIM_B * DIM_L)   // 20000
#define LDK      40                // padded LDS K-stride (bf16 elems)

// -------- async global -> LDS copy (gfx1250), guarded with fallback --------
#if defined(__HIP_DEVICE_COMPILE__) && defined(__gfx1250__) && \
    __has_builtin(__builtin_amdgcn_global_load_async_to_lds_b128) && \
    __has_builtin(__builtin_amdgcn_s_wait_asynccnt)
#define USE_ASYNC_LDS 1
#else
#define USE_ASYNC_LDS 0
#endif

#if USE_ASYNC_LDS
typedef int v4i_t __attribute__((ext_vector_type(4)));
typedef __attribute__((address_space(1))) v4i_t* as1_v4i_p;   // global int4*
typedef __attribute__((address_space(3))) v4i_t* as3_v4i_p;   // LDS    int4*
#endif

__device__ __forceinline__ void async_copy16(const void* g, void* l) {
#if USE_ASYNC_LDS
  __builtin_amdgcn_global_load_async_to_lds_b128(
      (as1_v4i_p)g, (as3_v4i_p)l, 0, 0);
#else
  (void)g; (void)l;
#endif
}
__device__ __forceinline__ void wait_async0() {
#if USE_ASYNC_LDS
  __builtin_amdgcn_s_wait_asynccnt(0);
#endif
}

// ---------------------------------------------------------------------------
__device__ __forceinline__ uint16_t f2bf_bits(float f) {
  union { float f; uint32_t u; } v; v.f = f;
  return (uint16_t)((v.u + 0x7fffu + ((v.u >> 16) & 1u)) >> 16);   // RNE
}
__device__ __forceinline__ bf16_t f2bf(float f) {
  union { uint16_t s; bf16_t b; } o; o.s = f2bf_bits(f);
  return o.b;
}
__device__ __forceinline__ float bf2f(bf16_t b) {
  union { uint16_t s; bf16_t b; } i; i.b = b;
  union { uint32_t u; float f; } o; o.u = ((uint32_t)i.s) << 16;
  return o.f;
}

union FragU { v16bf v; uint32_t u[8]; };

// Load a 16x32 bf16 A/B fragment from LDS (rows stored major, stride LDK).
// Per CDNA5 ISA: lanes 0-15 hold K {0..7,16..23}, lanes 16-31 hold K+8,
// each VGPR packs two consecutive K (one 32-bit LDS read per VGPR).
__device__ __forceinline__ v16bf load_frag(const bf16_t* lds, int rowBase, int lane) {
  const int half = lane >> 4;
  const bf16_t* p = lds + (rowBase + (lane & 15)) * LDK + half * 8;
  FragU f;
#pragma unroll
  for (int i = 0; i < 8; ++i) {
    int k = (i < 4) ? (2 * i) : (2 * (i - 4) + 16);
    f.u[i] = *(const uint32_t*)(p + k);
  }
  return f.v;
}

// ---------------------------------------------------------------------------
__global__ void f32_to_bf16_kernel(const float* __restrict__ in,
                                   bf16_t* __restrict__ out, int n) {
  int i = blockIdx.x * blockDim.x + threadIdx.x;
  if (i < n) out[i] = f2bf(in[i]);
}

// ---------------------------------------------------------------------------
// fc1: per batch, D[p, o] = sum_c feature[c, p] * w[o, c] + bias[o]
// A is K-major in memory (feature[c][p]) -> transpose while staging to LDS.
__global__ __launch_bounds__(256)
void fc1_gemm_kernel(const float* __restrict__ feat,   // [B][CIN][HW] f32
                     const bf16_t* __restrict__ Wc,    // [DIM_LOI][CIN] bf16
                     const float* __restrict__ bias,   // [DIM_LOI]
                     bf16_t* __restrict__ XT)          // [B][HW][DIM_LOI]
{
  __shared__ __align__(16) bf16_t lds_a[128 * LDK];
  __shared__ __align__(16) bf16_t lds_b[128 * LDK];

  const int tid  = threadIdx.x;
  const int lane = tid & 31;
  const int wave = tid >> 5;
  const int wm   = wave & 3;            // 4 waves in M -> 32 rows each
  const int wn   = wave >> 2;           // 2 waves in N -> 64 cols each
  const int m0   = blockIdx.x * 128;    // pixel tile
  const int b    = blockIdx.z;

  const float* fb = feat + (size_t)b * CIN * HW;
  bf16_t*      Db = XT   + (size_t)b * HW * DIM_LOI;

  v8f acc[2][4] = {};

  for (int kt = 0; kt < (CIN >> 5); ++kt) {
    const int k0 = kt << 5;
    __syncthreads();
    // stage A transposed: each thread handles two 4(m) x 2(k) micro-tiles,
    // packing K-pairs into u32 -> ds_store_b32 (fragment-aligned pairs).
#pragma unroll
    for (int it = 0; it < 2; ++it) {
      int chunk = tid + it * 256;               // 0..511
      int k  = (chunk >> 5) * 2;                // 0,2,..,30
      int cm = (chunk & 31) * 4;                // 0..124
      float4 d0 = *(const float4*)(fb + (size_t)(k0 + k + 0) * HW + m0 + cm);
      float4 d1 = *(const float4*)(fb + (size_t)(k0 + k + 1) * HW + m0 + cm);
      uint32_t* la = (uint32_t*)&lds_a[0];
      la[((cm + 0) * LDK + k) >> 1] = (uint32_t)f2bf_bits(d0.x) | ((uint32_t)f2bf_bits(d1.x) << 16);
      la[((cm + 1) * LDK + k) >> 1] = (uint32_t)f2bf_bits(d0.y) | ((uint32_t)f2bf_bits(d1.y) << 16);
      la[((cm + 2) * LDK + k) >> 1] = (uint32_t)f2bf_bits(d0.z) | ((uint32_t)f2bf_bits(d1.z) << 16);
      la[((cm + 3) * LDK + k) >> 1] = (uint32_t)f2bf_bits(d0.w) | ((uint32_t)f2bf_bits(d1.w) << 16);
    }
    // stage B: Wc[n][k] row-major -> lds_b[n][k], 16B chunks
#pragma unroll
    for (int it = 0; it < 2; ++it) {
      int chunk = tid + it * 256;               // 0..511
      int n  = chunk >> 2;                      // 0..127
      int ck = (chunk & 3) * 8;                 // 0,8,16,24
      uint4 e = *(const uint4*)(Wc + (size_t)n * CIN + k0 + ck);
      *(uint4*)(&lds_b[n * LDK + ck]) = e;
    }
    if (kt + 1 < (CIN >> 5))
      __builtin_prefetch(fb + (size_t)(k0 + 32 + (tid >> 5)) * HW + m0 + (tid & 31) * 4, 0, 1);
    __syncthreads();

    v16bf af[2], bfr[4];
#pragma unroll
    for (int i = 0; i < 2; ++i) af[i]  = load_frag(lds_a, wm * 32 + i * 16, lane);
#pragma unroll
    for (int j = 0; j < 4; ++j) bfr[j] = load_frag(lds_b, wn * 64 + j * 16, lane);
#pragma unroll
    for (int i = 0; i < 2; ++i)
#pragma unroll
      for (int j = 0; j < 4; ++j)
        acc[i][j] = __builtin_amdgcn_wmma_f32_16x16x32_bf16(
            false, af[i], false, bfr[j], (short)0, acc[i][j], false, false);
  }

  // epilogue: D[p][o] bf16, C/D layout: VGPR r -> M = r + half*8, N = lane&15
  const int half = lane >> 4;
  const int ncol = lane & 15;
#pragma unroll
  for (int i = 0; i < 2; ++i)
#pragma unroll
    for (int j = 0; j < 4; ++j) {
      int n = wn * 64 + j * 16 + ncol;
      float bv = bias[n];
#pragma unroll
      for (int r = 0; r < 8; ++r) {
        int m = m0 + wm * 32 + i * 16 + half * 8 + r;
        Db[(size_t)m * DIM_LOI + n] = f2bf(acc[i][j][r] + bv);
      }
    }
}

// ---------------------------------------------------------------------------
// Bilinear sample 32 points per line from x_t[b][p][128], maxpool groups of 4.
__global__ __launch_bounds__(128)
void loi_gather_kernel(const float* __restrict__ lines,   // [B][L][2][2]
                       const bf16_t* __restrict__ xt,     // [B][HW][128]
                       bf16_t* __restrict__ feats)        // [B*L][1024]
{
  const int l = blockIdx.x;
  const int b = blockIdx.y;
  const int c = threadIdx.x;                   // channel 0..127
  const float* ln = lines + ((size_t)b * DIM_L + l) * 4;
  const float r0 = ln[0], c0 = ln[1], r1 = ln[2], c1 = ln[3];
  const bf16_t* xb = xt + (size_t)b * HW * DIM_LOI;
  bf16_t* fr = feats + ((size_t)b * DIM_L + l) * (DIM_LOI * 8) + c * 8;

  float mx = -3.4e38f;
#pragma unroll 4
  for (int j = 0; j < 32; ++j) {
    float lam = (float)j * (1.0f / 31.0f);
    float px = r0 * lam + r1 * (1.0f - lam) - 0.5f;   // H index
    float py = c0 * lam + c1 * (1.0f - lam) - 0.5f;   // W index
    float fx0 = fminf(fmaxf(floorf(px), 0.0f), 255.0f);
    float fy0 = fminf(fmaxf(floorf(py), 0.0f), 255.0f);
    float fx1 = fminf(fx0 + 1.0f, 255.0f);
    float fy1 = fminf(fy0 + 1.0f, 255.0f);
    int ix0 = (int)fx0, iy0 = (int)fy0, ix1 = (int)fx1, iy1 = (int)fy1;
    float w00 = (fx1 - px) * (fy1 - py);
    float w10 = (px - fx0) * (fy1 - py);
    float w01 = (fx1 - px) * (py - fy0);
    float w11 = (px - fx0) * (py - fy0);
    float v = bf2f(xb[(size_t)(ix0 * 256 + iy0) * DIM_LOI + c]) * w00
            + bf2f(xb[(size_t)(ix1 * 256 + iy0) * DIM_LOI + c]) * w10
            + bf2f(xb[(size_t)(ix0 * 256 + iy1) * DIM_LOI + c]) * w01
            + bf2f(xb[(size_t)(ix1 * 256 + iy1) * DIM_LOI + c]) * w11;
    mx = fmaxf(mx, v);
    if ((j & 3) == 3) { fr[j >> 2] = f2bf(mx); mx = -3.4e38f; }
  }
}

// ---------------------------------------------------------------------------
// Issue one 128x32 bf16 A-tile + B-tile as async global->LDS b128 copies.
__device__ __forceinline__ void mlp_issue_tile(const bf16_t* A, const bf16_t* W,
                                               bf16_t* la, bf16_t* lb,
                                               int tid, int m0, int n0,
                                               int M, int K, int k0)
{
#pragma unroll
  for (int it = 0; it < 2; ++it) {
    int chunk = tid + it * 256;               // 0..511
    int row = chunk >> 2;                     // 0..127
    int ck  = (chunk & 3) * 8;                // 0,8,16,24
    int mg = m0 + row; if (mg >= M) mg = M - 1;
    async_copy16(A + (size_t)mg * K + k0 + ck,        la + row * LDK + ck);
    async_copy16(W + (size_t)(n0 + row) * K + k0 + ck, lb + row * LDK + ck);
  }
}

// MLP GEMM: D = act(A[M,K] @ W[N,K]^T + bias), bf16 in/out, f32 accumulate.
__global__ __launch_bounds__(256)
void mlp_gemm_kernel(const bf16_t* __restrict__ A, const bf16_t* __restrict__ W,
                     const float* __restrict__ bias, bf16_t* __restrict__ D,
                     int M, int N, int K, int relu)
{
#if USE_ASYNC_LDS
  __shared__ __align__(16) bf16_t lds_a[2][128 * LDK];
  __shared__ __align__(16) bf16_t lds_b[2][128 * LDK];
#else
  __shared__ __align__(16) bf16_t lds_a[1][128 * LDK];
  __shared__ __align__(16) bf16_t lds_b[1][128 * LDK];
#endif

  const int tid  = threadIdx.x;
  const int lane = tid & 31;
  const int wave = tid >> 5;
  const int wm   = wave & 3;
  const int wn   = wave >> 2;
  const int m0   = blockIdx.x * 128;
  const int n0   = blockIdx.y * 128;

  v8f acc[2][4] = {};
  const int nK = K >> 5;

#if USE_ASYNC_LDS
  // Double-buffered async pipeline: tile kt lives in buffer kt&1.
  mlp_issue_tile(A, W, &lds_a[0][0], &lds_b[0][0], tid, m0, n0, M, K, 0);
  for (int kt = 0; kt < nK; ++kt) {
    wait_async0();          // my async writes (into buffer kt&1) complete
    __syncthreads();        // everyone's writes complete; prior reads done
    const int cur = kt & 1;
    if (kt + 1 < nK)
      mlp_issue_tile(A, W, &lds_a[1 - cur][0], &lds_b[1 - cur][0],
                     tid, m0, n0, M, K, (kt + 1) << 5);
    v16bf af[2], bfr[4];
#pragma unroll
    for (int i = 0; i < 2; ++i) af[i]  = load_frag(&lds_a[cur][0], wm * 32 + i * 16, lane);
#pragma unroll
    for (int j = 0; j < 4; ++j) bfr[j] = load_frag(&lds_b[cur][0], wn * 64 + j * 16, lane);
#pragma unroll
    for (int i = 0; i < 2; ++i)
#pragma unroll
      for (int j = 0; j < 4; ++j)
        acc[i][j] = __builtin_amdgcn_wmma_f32_16x16x32_bf16(
            false, af[i], false, bfr[j], (short)0, acc[i][j], false, false);
  }
#else
  for (int kt = 0; kt < nK; ++kt) {
    const int k0 = kt << 5;
    __syncthreads();
#pragma unroll
    for (int it = 0; it < 2; ++it) {
      int chunk = tid + it * 256;
      int row = chunk >> 2;
      int ck  = (chunk & 3) * 8;
      int mg = m0 + row; if (mg >= M) mg = M - 1;
      *(uint4*)(&lds_a[0][row * LDK + ck]) = *(const uint4*)(A + (size_t)mg * K + k0 + ck);
      *(uint4*)(&lds_b[0][row * LDK + ck]) = *(const uint4*)(W + (size_t)(n0 + row) * K + k0 + ck);
    }
    if (kt + 1 < nK) {
      int pm = m0 + (tid >> 1); if (pm >= M) pm = M - 1;
      __builtin_prefetch(A + (size_t)pm * K + k0 + 32, 0, 1);
      __builtin_prefetch(W + (size_t)(n0 + (tid >> 1)) * K + k0 + 32, 0, 1);
    }
    __syncthreads();
    v16bf af[2], bfr[4];
#pragma unroll
    for (int i = 0; i < 2; ++i) af[i]  = load_frag(&lds_a[0][0], wm * 32 + i * 16, lane);
#pragma unroll
    for (int j = 0; j < 4; ++j) bfr[j] = load_frag(&lds_b[0][0], wn * 64 + j * 16, lane);
#pragma unroll
    for (int i = 0; i < 2; ++i)
#pragma unroll
      for (int j = 0; j < 4; ++j)
        acc[i][j] = __builtin_amdgcn_wmma_f32_16x16x32_bf16(
            false, af[i], false, bfr[j], (short)0, acc[i][j], false, false);
  }
#endif

  const int half = lane >> 4;
  const int ncol = lane & 15;
#pragma unroll
  for (int i = 0; i < 2; ++i)
#pragma unroll
    for (int j = 0; j < 4; ++j) {
      int n = n0 + wn * 64 + j * 16 + ncol;
      float bv = bias[n];
#pragma unroll
      for (int r = 0; r < 8; ++r) {
        int m = m0 + wm * 32 + i * 16 + half * 8 + r;
        float v = acc[i][j][r] + bv;
        if (relu) v = fmaxf(v, 0.0f);
        if (m < M) D[(size_t)m * N + n] = f2bf(v);
      }
    }
}

// ---------------------------------------------------------------------------
// Final 4-logit head: out[m, o] = h2[m,:] . w3[o,:] + b3[o]   (fp32 out)
__global__ __launch_bounds__(128)
void final_fc_kernel(const bf16_t* __restrict__ h2, const float* __restrict__ w3,
                     const float* __restrict__ b3, float* __restrict__ out)
{
  const int row  = blockIdx.x;
  const int o    = threadIdx.x >> 5;        // 4 waves -> 4 outputs
  const int lane = threadIdx.x & 31;
  const bf16_t* hr = h2 + (size_t)row * DIM_FC;
  const float*  wr = w3 + (size_t)o * DIM_FC;
  float s = 0.0f;
#pragma unroll 4
  for (int i = lane; i < DIM_FC; i += 32) s += bf2f(hr[i]) * wr[i];
#pragma unroll
  for (int off = 16; off > 0; off >>= 1) s += __shfl_down(s, off, 32);
  if (lane == 0) out[(size_t)row * 4 + o] = s + b3[o];
}

// ---------------------------------------------------------------------------
extern "C" void kernel_launch(void* const* d_in, const int* in_sizes, int n_in,
                              void* d_out, int out_size, void* d_ws, size_t ws_size,
                              hipStream_t stream)
{
  const float* feature = (const float*)d_in[0];
  const float* lines   = (const float*)d_in[1];
  const float* w_fc1   = (const float*)d_in[2];
  const float* b_fc1   = (const float*)d_in[3];
  const float* w1      = (const float*)d_in[4];
  const float* b1      = (const float*)d_in[5];
  const float* w2      = (const float*)d_in[6];
  const float* b2      = (const float*)d_in[7];
  const float* w3      = (const float*)d_in[8];
  const float* b3      = (const float*)d_in[9];
  float* out = (float*)d_out;

  // workspace layout (bytes):
  //   [0, 64K)            w_fc1 bf16
  //   [64K, 64K+2M)       w1 bf16
  //   [64K+2M, 64K+4M)    w2 bf16
  //   [XT, XT+64M)        x_t bf16  -> reused as h1 (40MB) after gather
  //   [FT, FT+40M)        feats bf16 -> reused as h2 after MLP1
  char* ws = (char*)d_ws;
  bf16_t* wfc1b = (bf16_t*)(ws);
  bf16_t* w1b   = (bf16_t*)(ws + (64u << 10));
  bf16_t* w2b   = (bf16_t*)(ws + (64u << 10) + (2u << 20));
  size_t xt_off = (64u << 10) + (4u << 20);
  bf16_t* xt    = (bf16_t*)(ws + xt_off);
  bf16_t* h1    = xt;                                          // reuse
  size_t ft_off = xt_off + (size_t)DIM_B * HW * DIM_LOI * 2;
  bf16_t* feats = (bf16_t*)(ws + ft_off);
  bf16_t* h2    = feats;                                       // reuse

  // 1) weight conversion
  f32_to_bf16_kernel<<<(DIM_LOI * CIN + 255) / 256, 256, 0, stream>>>(w_fc1, wfc1b, DIM_LOI * CIN);
  f32_to_bf16_kernel<<<(DIM_FC * DIM_FC + 255) / 256, 256, 0, stream>>>(w1, w1b, DIM_FC * DIM_FC);
  f32_to_bf16_kernel<<<(DIM_FC * DIM_FC + 255) / 256, 256, 0, stream>>>(w2, w2b, DIM_FC * DIM_FC);

  // 2) fc1 GEMM -> x_t [B, HW, 128] bf16
  dim3 g1(HW / 128, 1, DIM_B);
  fc1_gemm_kernel<<<g1, 256, 0, stream>>>(feature, wfc1b, b_fc1, xt);

  // 3) line sampling + maxpool -> feats [20000, 1024] bf16
  dim3 g2(DIM_L, DIM_B, 1);
  loi_gather_kernel<<<g2, 128, 0, stream>>>(lines, xt, feats);

  // 4) MLP (double-buffered async-to-LDS WMMA GEMMs)
  dim3 g3((NLINES + 127) / 128, DIM_FC / 128, 1);
  mlp_gemm_kernel<<<g3, 256, 0, stream>>>(feats, w1b, b1, h1, NLINES, DIM_FC, DIM_FC, 1);
  mlp_gemm_kernel<<<g3, 256, 0, stream>>>(h1, w2b, b2, h2, NLINES, DIM_FC, DIM_FC, 1);

  // 5) 4-logit head -> d_out fp32 [20000, 4]
  final_fc_kernel<<<NLINES, 128, 0, stream>>>(h2, w3, b3, out);
}
